// PFCBlock_2113123909734
// MI455X (gfx1250) — matrix operand order repaired and checked by reading, not compiled, not run on hardware
//
#include <hip/hip_runtime.h>
#include <math.h>

// ---------------------------------------------------------------------------
// Problem dims (from reference setup_inputs)
// ---------------------------------------------------------------------------
#define XD    784      // x_dim (K of the setup GEMMs)
#define KDIM  1024     // number of components
#define BATCH 2048
#define YDIM  10

#define N_ITERS_TOTAL 60   // 50 no-grad + 10 grad (forward math identical)

// LDS tile stride (32 K-chunk + 8 pad bf16 -> 80B rows, 16B aligned)
#define LDSK 40

// fista tile: block 64(M) x 128(N), 4 waves (2x2), wave tile 32x64
#define BM 64
#define BN 128

typedef __attribute__((ext_vector_type(16))) __bf16 v16bf;
typedef __attribute__((ext_vector_type(8)))  __bf16 v8bf;
typedef __attribute__((ext_vector_type(8)))  float  v8f;

#define AS1 __attribute__((address_space(1)))
#define AS3 __attribute__((address_space(3)))

// gcc-style int4 to match the async-LDS builtin's parameter type exactly
typedef int async_i4 __attribute__((vector_size(16)));

// ---------------------------------------------------------------------------
// Async global->LDS copy (gfx1250 GLOBAL_LOAD_ASYNC_TO_LDS_B128, ASYNCcnt).
// Guarded: falls back to a synchronous copy if the builtin is absent.
// ---------------------------------------------------------------------------
#if __has_builtin(__builtin_amdgcn_global_load_async_to_lds_b128)
#define HAVE_ASYNC_LDS 1
#else
#define HAVE_ASYNC_LDS 0
#endif

__device__ __forceinline__ void async_cp16(const __bf16* g, __bf16* l) {
#if HAVE_ASYNC_LDS
    __builtin_amdgcn_global_load_async_to_lds_b128(
        (AS1 async_i4*)g, (AS3 async_i4*)l, 0, 0);
#else
    *(uint4*)l = *(const uint4*)g;
#endif
}

#if __has_builtin(__builtin_amdgcn_s_wait_asynccnt)
#define WAIT_ASYNC() __builtin_amdgcn_s_wait_asynccnt(0)
#else
#define WAIT_ASYNC() asm volatile("s_wait_asynccnt 0x0" ::: "memory")
#endif

// ---------------------------------------------------------------------------
// Fragment loads from LDS (ISA 7.12.2 layouts, wave32)
// A (16x32, MxK):  lane l holds row M=(l&15); K = h*8+{0..7} and 16+h*8+{0..7}
// B (32x16, KxN):  lane l holds col N=(l&15); K = h*8+{0..7} and 16+h*8+{0..7}
// ---------------------------------------------------------------------------
__device__ __forceinline__ v16bf ld_frag_a(const __bf16* tile, int row, int half) {
    const v8bf lo = *(const v8bf*)(tile + row * LDSK + half * 8);
    const v8bf hi = *(const v8bf*)(tile + row * LDSK + 16 + half * 8);
    return __builtin_shufflevector(lo, hi, 0,1,2,3,4,5,6,7,8,9,10,11,12,13,14,15);
}
__device__ __forceinline__ v16bf ld_frag_b(const __bf16* tile, int col, int half) {
    const v8bf lo = *(const v8bf*)(tile + col * LDSK + half * 8);
    const v8bf hi = *(const v8bf*)(tile + col * LDSK + 16 + half * 8);
    return __builtin_shufflevector(lo, hi, 0,1,2,3,4,5,6,7,8,9,10,11,12,13,14,15);
}

// ---------------------------------------------------------------------------
// Setup GEMM:  C[M,N] = A^T B,  A:[K,M] row-major (lda), B:[K,N] row-major (ldb)
// f32 inputs converted to bf16 while staging. Writes f32 C and optional bf16 C.
// Block: 128 threads (4 waves, 2x2), tile 64x64, K-step 32 (zero-padded).
// ---------------------------------------------------------------------------
__global__ __launch_bounds__(128)
void gemm_atb_bf16(const float* __restrict__ A, const float* __restrict__ B,
                   float* __restrict__ Cf, __bf16* __restrict__ Cbf,
                   int K, int M, int N, int lda, int ldb) {
    __shared__ __bf16 Atile[64 * LDSK];
    __shared__ __bf16 Btile[64 * LDSK];

    const int tid  = threadIdx.x;
    const int wave = tid >> 5;
    const int lane = tid & 31;
    const int half = lane >> 4;
    const int l15  = lane & 15;
    const int mW   = (wave >> 1) * 32;
    const int nW   = (wave & 1) * 32;

    const int nTile = blockIdx.x * 64;
    const int mTile = blockIdx.y * 64;

    v8f acc[2][2];
    #pragma unroll
    for (int i = 0; i < 2; ++i)
        #pragma unroll
        for (int j = 0; j < 2; ++j) acc[i][j] = (v8f)0.0f;

    const int kSteps = (K + 31) / 32;
    for (int ks = 0; ks < kSteps; ++ks) {
        const int kTile = ks * 32;
        #pragma unroll
        for (int e = 0; e < 16; ++e) {
            const int lin = tid + e * 128;       // 0..2047
            const int m   = lin & 63;
            const int kk  = lin >> 6;
            const int kg  = kTile + kk;
            float av = 0.0f, bv = 0.0f;
            if (kg < K) {
                av = A[(size_t)kg * lda + (mTile + m)];
                bv = B[(size_t)kg * ldb + (nTile + m)];
            }
            Atile[m * LDSK + kk] = (__bf16)av;
            Btile[m * LDSK + kk] = (__bf16)bv;
        }
        __syncthreads();

        v16bf af[2], bg[2];
        af[0] = ld_frag_a(Atile, mW + l15,      half);
        af[1] = ld_frag_a(Atile, mW + 16 + l15, half);
        bg[0] = ld_frag_b(Btile, nW + l15,      half);
        bg[1] = ld_frag_b(Btile, nW + 16 + l15, half);

        #pragma unroll
        for (int i = 0; i < 2; ++i)
            #pragma unroll
            for (int j = 0; j < 2; ++j)
                acc[i][j] = __builtin_amdgcn_wmma_f32_16x16x32_bf16(
                    false, af[i], false, bg[j], (short)0, acc[i][j], false, false);
        __syncthreads();
    }

    #pragma unroll
    for (int i = 0; i < 2; ++i)
        #pragma unroll
        for (int j = 0; j < 2; ++j)
            #pragma unroll
            for (int r = 0; r < 8; ++r) {
                const int m = mTile + mW + i * 16 + half * 8 + r;
                const int n = nTile + nW + j * 16 + l15;
                const size_t idx = (size_t)m * N + n;
                const float v = acc[i][j][r];
                Cf[idx] = v;
                if (Cbf) Cbf[idx] = (__bf16)v;
            }
}

// ---------------------------------------------------------------------------
// Power iteration on WtW (f32, 1024x1024), single 1024-thread workgroup.
// ---------------------------------------------------------------------------
__global__ __launch_bounds__(1024)
void power_iter_kernel(const float* __restrict__ WtW, float* __restrict__ invL) {
    __shared__ float v[KDIM];
    __shared__ float red[KDIM];
    const int t = threadIdx.x;

    v[t] = 0.03125f;   // 1/sqrt(1024)
    __syncthreads();

    const float4* row4 = (const float4*)(WtW + (size_t)t * KDIM);
    float w = 0.0f;
    for (int it = 0; it < 30; ++it) {
        w = 0.0f;
        #pragma unroll 4
        for (int k = 0; k < KDIM; k += 4) {
            const float4 r = row4[k >> 2];
            w += r.x * v[k] + r.y * v[k + 1] + r.z * v[k + 2] + r.w * v[k + 3];
        }
        red[t] = w * w;
        __syncthreads();
        for (int s = 512; s > 0; s >>= 1) {
            if (t < s) red[t] += red[t + s];
            __syncthreads();
        }
        const float nrm = sqrtf(red[0]) + 1e-12f;
        __syncthreads();
        v[t] = w / nrm;
        __syncthreads();
    }
    w = 0.0f;
    #pragma unroll 4
    for (int k = 0; k < KDIM; k += 4) {
        const float4 r = row4[k >> 2];
        w += r.x * v[k] + r.y * v[k + 1] + r.z * v[k + 2] + r.w * v[k + 3];
    }
    red[t] = v[t] * w;
    __syncthreads();
    for (int s = 512; s > 0; s >>= 1) {
        if (t < s) red[t] += red[t + s];
        __syncthreads();
    }
    if (t == 0) *invL = 1.0f / (red[0] + 1e-20f);
}

// ---------------------------------------------------------------------------
// State init: H = Y = h_init, Ybf = bf16(h_init)
// ---------------------------------------------------------------------------
__global__ __launch_bounds__(256)
void init_state(const float* __restrict__ h0, float* __restrict__ H,
                float* __restrict__ Y, __bf16* __restrict__ Ybf) {
    const size_t i = (size_t)blockIdx.x * 256 + threadIdx.x;
    const float v = h0[i];
    H[i] = v; Y[i] = v; Ybf[i] = (__bf16)v;
}

// ---------------------------------------------------------------------------
// Fused FISTA step (dominant kernel, launched 60x):
//   G = Y(bf16) @ WtW(bf16), f32 accum; epilogue does prox + momentum.
// Block 128 threads / 4 waves; block tile 64x128; wave tile 32x64 -> 8 WMMA
// per k-step per wave. LDS double-buffered; next k-chunk staged with
// GLOBAL_LOAD_ASYNC_TO_LDS_B128 while current chunk's WMMAs execute.
// WtW symmetric -> B-fragment columns read as contiguous WtW rows.
// ---------------------------------------------------------------------------
__global__ __launch_bounds__(128)
void fista_step(const __bf16* __restrict__ YbfIn, const __bf16* __restrict__ Wbf,
                const float* __restrict__ XtW, float* __restrict__ H,
                float* __restrict__ Y, __bf16* __restrict__ YbfOut,
                const float* __restrict__ invLp, float coef) {
    __shared__ __bf16 Atile[2][BM * LDSK];
    __shared__ __bf16 Btile[2][BN * LDSK];

    const int tid  = threadIdx.x;
    const int wave = tid >> 5;
    const int lane = tid & 31;
    const int half = lane >> 4;
    const int l15  = lane & 15;
    const int mW   = (wave >> 1) * 32;   // waves 2x2 over 64x128
    const int nW   = (wave & 1) * 64;

    const int nTile = blockIdx.x * BN;
    const int mTile = blockIdx.y * BM;

    // prefetch epilogue state (L2 hint)
    {
        const size_t pidx = (size_t)(mTile + mW + half * 8) * KDIM + nTile + nW + l15;
        __builtin_prefetch(XtW + pidx, 0, 1);
        __builtin_prefetch(Y + pidx, 1, 1);
        __builtin_prefetch(H + pidx, 1, 1);
    }

    // staging roles: A row = tid>>1 (0..63), 32B half (tid&1); B row = tid, 64B
    const int ar = tid >> 1;
    const int ah = tid & 1;

    auto stage = [&](int buf, int kTile) {
        const __bf16* ga = YbfIn + (size_t)(mTile + ar) * KDIM + kTile + ah * 16;
        __bf16*       la = &Atile[buf][ar * LDSK + ah * 16];
        async_cp16(ga,     la);
        async_cp16(ga + 8, la + 8);
        const __bf16* gb = Wbf + (size_t)(nTile + tid) * KDIM + kTile;
        __bf16*       lb = &Btile[buf][tid * LDSK];
        async_cp16(gb,      lb);
        async_cp16(gb + 8,  lb + 8);
        async_cp16(gb + 16, lb + 16);
        async_cp16(gb + 24, lb + 24);
    };

    v8f acc[2][4];
    #pragma unroll
    for (int i = 0; i < 2; ++i)
        #pragma unroll
        for (int j = 0; j < 4; ++j) acc[i][j] = (v8f)0.0f;

    stage(0, 0);
    WAIT_ASYNC();
    __syncthreads();

    const int KSTEPS = KDIM / 32;
    for (int ks = 0; ks < KSTEPS; ++ks) {
        const int cur = ks & 1;
        if (ks + 1 < KSTEPS) stage(cur ^ 1, (ks + 1) * 32);

        const __bf16* At = Atile[cur];
        const __bf16* Bt = Btile[cur];
        v16bf af[2], bg[4];
        #pragma unroll
        for (int i = 0; i < 2; ++i) af[i] = ld_frag_a(At, mW + i * 16 + l15, half);
        #pragma unroll
        for (int j = 0; j < 4; ++j) bg[j] = ld_frag_b(Bt, nW + j * 16 + l15, half);

        #pragma unroll
        for (int i = 0; i < 2; ++i)
            #pragma unroll
            for (int j = 0; j < 4; ++j)
                acc[i][j] = __builtin_amdgcn_wmma_f32_16x16x32_bf16(
                    false, af[i], false, bg[j], (short)0, acc[i][j], false, false);

        WAIT_ASYNC();
        __syncthreads();
    }

    const float invL = *invLp;
    #pragma unroll
    for (int i = 0; i < 2; ++i)
        #pragma unroll
        for (int j = 0; j < 4; ++j)
            #pragma unroll
            for (int r = 0; r < 8; ++r) {
                const int m = mTile + mW + i * 16 + half * 8 + r;
                const int n = nTile + nW + j * 16 + l15;
                const size_t idx = (size_t)m * KDIM + n;
                const float grad = acc[i][j][r] - XtW[idx];
                const float yv = Y[idx];
                const float hv = H[idx];
                const float hn = fmaxf(yv - grad * invL, 0.0f);
                const float yn = hn + coef * (hn - hv);
                H[idx] = hn;
                Y[idx] = yn;
                YbfOut[idx] = (__bf16)yn;
            }
}

// ---------------------------------------------------------------------------
// Finisher: out[i,b] = sum_j Wy[i,j] * H[b,j]   ([10,2048], ~42 MFLOP)
// ---------------------------------------------------------------------------
__global__ __launch_bounds__(256)
void final_gemm(const float* __restrict__ Wy, const float* __restrict__ H,
                float* __restrict__ out) {
    const int gid = blockIdx.x * 256 + threadIdx.x;
    if (gid >= YDIM * BATCH) return;
    const int i = gid >> 11;       // / 2048
    const int b = gid & 2047;
    const float4* wy4 = (const float4*)(Wy + (size_t)i * KDIM);
    const float4* h4  = (const float4*)(H + (size_t)b * KDIM);
    float acc = 0.0f;
    #pragma unroll 4
    for (int j = 0; j < KDIM / 4; ++j) {
        const float4 w = wy4[j];
        const float4 h = h4[j];
        acc += w.x * h.x + w.y * h.y + w.z * h.z + w.w * h.w;
    }
    out[(size_t)i * BATCH + b] = acc;
}

// ---------------------------------------------------------------------------
// Host launcher
// ---------------------------------------------------------------------------
extern "C" void kernel_launch(void* const* d_in, const int* in_sizes, int n_in,
                              void* d_out, int out_size, void* d_ws, size_t ws_size,
                              hipStream_t stream) {
    const float* x  = (const float*)d_in[0];   // [784, 2048]
    const float* Wx = (const float*)d_in[1];   // [784, 1024]
    const float* Wy = (const float*)d_in[2];   // [10, 1024]
    const float* h0 = (const float*)d_in[3];   // [2048, 1024]
    float* out = (float*)d_out;                // [10, 2048]

    char* ws = (char*)d_ws;
    const size_t MB = 1024ull * 1024ull;
    float*  WtWf  = (float*)(ws + 0 * MB);         // 4 MB
    float*  XtW   = (float*)(ws + 4 * MB);         // 8 MB
    float*  H     = (float*)(ws + 12 * MB);        // 8 MB
    float*  Y     = (float*)(ws + 20 * MB);        // 8 MB
    __bf16* WtWb  = (__bf16*)(ws + 28 * MB);       // 2 MB
    __bf16* Ybf0  = (__bf16*)(ws + 30 * MB);       // 4 MB
    __bf16* Ybf1  = (__bf16*)(ws + 34 * MB);       // 4 MB
    float*  invLp = (float*)(ws + 38 * MB);        // 4 B

    // momentum coefficients (t restarts at iter 50; coef[49]=0 resets y<-h)
    float coefs[N_ITERS_TOTAL];
    double t = 1.0;
    for (int i = 0; i < N_ITERS_TOTAL; ++i) {
        if (i == 50) t = 1.0;
        const double tn = 0.5 * (1.0 + sqrt(1.0 + 4.0 * t * t));
        coefs[i] = (float)((t - 1.0) / tn);
        t = tn;
    }
    coefs[49] = 0.0f;

    // 1) WtW = Wx^T Wx  (f32 + bf16 copy)
    gemm_atb_bf16<<<dim3(KDIM / 64, KDIM / 64), 128, 0, stream>>>(
        Wx, Wx, WtWf, WtWb, XD, KDIM, KDIM, KDIM, KDIM);
    // 2) XtW = x^T Wx   (f32 only)
    gemm_atb_bf16<<<dim3(KDIM / 64, BATCH / 64), 128, 0, stream>>>(
        x, Wx, XtW, (__bf16*)nullptr, XD, BATCH, KDIM, BATCH, KDIM);
    // 3) invL via power iteration
    power_iter_kernel<<<1, 1024, 0, stream>>>(WtWf, invLp);
    // 4) H = Y = h_init, Ybf0 = bf16(h_init)
    init_state<<<(BATCH * KDIM) / 256, 256, 0, stream>>>(h0, H, Y, Ybf0);
    // 5) 60 fused FISTA steps (bf16 Y double-buffered)
    for (int i = 0; i < N_ITERS_TOTAL; ++i) {
        const __bf16* yin  = (i & 1) ? Ybf1 : Ybf0;
        __bf16*       yout = (i & 1) ? Ybf0 : Ybf1;
        fista_step<<<dim3(KDIM / BN, BATCH / BM), 128, 0, stream>>>(
            yin, WtWb, XtW, H, Y, yout, invLp, coefs[i]);
    }
    // 6) y_pred = Wy @ H^T
    final_gemm<<<(YDIM * BATCH + 255) / 256, 256, 0, stream>>>(Wy, H, out);
}
